// RNNLayer_89103391523029
// MI455X (gfx1250) — matrix operand order repaired and checked by reading, not compiled, not run on hardware
//
#include <hip/hip_runtime.h>
#include <math.h>

// Problem constants (from reference)
#define T_STEPS 512
#define NB      128      // batch
#define DIN     512      // input dim
#define HID     512      // hidden dim
#define NL      2        // layers
#define G4      (4*HID)  // 2048 gate rows
#define EPS     1e-5f
#define NBLOCKS 32       // persistent grid size (co-resident)
#define NBH     (NB*HID) // elements of one layer's h state

typedef __attribute__((ext_vector_type(16))) _Float16 v16h;
typedef __attribute__((ext_vector_type(8)))  _Float16 v8h;
typedef __attribute__((ext_vector_type(8)))  float    v8f;

__device__ __forceinline__ float sigmoidf_(float x) {
    return 1.0f / (1.0f + __expf(-x));
}

// ---- A-fragment loaders (16x32, row stride 512) ----
// ISA layout: lane L -> row L&15,
//   halves 0..7  = K in [k0 + (L>>4)*8,      +8)
//   halves 8..15 = K in [k0 + 16 + (L>>4)*8, +8)

__device__ __forceinline__ v16h load_a_f16(const _Float16* __restrict__ src,
                                           int row, int k0, int lane) {
    const _Float16* p = src + row * 512 + k0 + ((lane >> 4) << 3);
    const v8h lo = *(const v8h*)(p);
    const v8h hi = *(const v8h*)(p + 16);
    v16h a;
#pragma unroll
    for (int i = 0; i < 8; ++i) { a[i] = lo[i]; a[8 + i] = hi[i]; }
    return a;
}

__device__ __forceinline__ v16h load_a_f16_scaled(const _Float16* __restrict__ src,
                                                  int row, int k0, int lane,
                                                  _Float16 mh) {
    const _Float16* p = src + row * 512 + k0 + ((lane >> 4) << 3);
    const v8h lo = *(const v8h*)(p);
    const v8h hi = *(const v8h*)(p + 16);
    v16h a;
#pragma unroll
    for (int i = 0; i < 8; ++i) { a[i] = lo[i] * mh; a[8 + i] = hi[i] * mh; }
    return a;
}

__device__ __forceinline__ v16h load_a_f32(const float* __restrict__ src,
                                           int row, int k0, int lane) {
    const float* p = src + row * 512 + k0 + ((lane >> 4) << 3);
    float f[16];
    *(float4*)(f + 0)  = *(const float4*)(p + 0);
    *(float4*)(f + 4)  = *(const float4*)(p + 4);
    *(float4*)(f + 8)  = *(const float4*)(p + 16);
    *(float4*)(f + 12) = *(const float4*)(p + 20);
    v16h a;
#pragma unroll
    for (int i = 0; i < 16; ++i) a[i] = (_Float16)f[i];
    return a;
}

// Device-scope generation barrier across NBLOCKS co-resident workgroups.
// gen release-store orders the cnt reset; spinners acquire-load gen.
__device__ __forceinline__ void grid_barrier(unsigned* cnt, unsigned* gen) {
    __threadfence();   // make this block's stores device-visible
    __syncthreads();
    if (threadIdx.x == 0) {
        const unsigned g = __hip_atomic_load(gen, __ATOMIC_RELAXED,
                                             __HIP_MEMORY_SCOPE_AGENT);
        const unsigned arrived = __hip_atomic_fetch_add(
            cnt, 1u, __ATOMIC_ACQ_REL, __HIP_MEMORY_SCOPE_AGENT);
        if (arrived == (unsigned)(NBLOCKS - 1)) {
            __hip_atomic_store(cnt, 0u, __ATOMIC_RELAXED,
                               __HIP_MEMORY_SCOPE_AGENT);
            __hip_atomic_store(gen, g + 1u, __ATOMIC_RELEASE,
                               __HIP_MEMORY_SCOPE_AGENT);
        } else {
            while (__hip_atomic_load(gen, __ATOMIC_ACQUIRE,
                                     __HIP_MEMORY_SCOPE_AGENT) == g) {
                __builtin_amdgcn_s_sleep(1);
            }
        }
    }
    __syncthreads();
    __threadfence();   // invalidate stale near-cache lines before re-reading
}

// One LSTM layer, this wave's 16(batch)x16(hidden) tile, all 4 gates.
// z = xin @ Wih^T + (m*hprev) @ Whh^T + b ; gate math; c/h update.
// NOTE: K loops use unroll 4 — full unroll overflows the register file and
// spills B fragments to scratch (observed: scratch_load_b128 reloads +
// s_set_vgpr_msb in the hot loop).
template <bool AF16>
__device__ __forceinline__ void lstm_layer_tile(
    const void*     __restrict__ xin_,   // (128,512) f16 or fp32
    const _Float16* __restrict__ hprev,  // (128,512) f16
    _Float16*       __restrict__ hnext,  // (128,512) f16
    float*          __restrict__ cstate, // fp32, row stride 1024 (in hxs_out)
    const _Float16* __restrict__ Wih,    // (2048,512) f16 row-major
    const _Float16* __restrict__ Whh,    // (2048,512) f16 row-major
    const float*    __restrict__ bias,   // (2048,)
    const float*    __restrict__ mask_t, // (128,)
    float*          __restrict__ out_t,  // fp32 (128,512) or nullptr
    float*          __restrict__ hdump,  // fp32, row stride 1024, or nullptr
    int lane, int bt, int ht)
{
    const int rowA = bt * 16 + (lane & 15);     // A-fragment row (batch)
    const int col  = ht * 16 + (lane & 15);     // C/D column (hidden unit)
    const _Float16 mh = (_Float16)mask_t[rowA]; // mask is exactly 0.0 or 1.0

    v8f acc[4];
#pragma unroll
    for (int q = 0; q < 4; ++q) {
        const float bq = bias[q * HID + col];
#pragma unroll
        for (int v = 0; v < 8; ++v) acc[q][v] = bq;
    }

    const int bOff = ((lane >> 4) << 3);  // B-fragment K offset within 32-chunk

    // ---- input contribution: xin @ Wih^T ----
#pragma unroll 4
    for (int kk = 0; kk < 16; ++kk) {
        const int k0 = kk * 32;
        v16h a;
        if (AF16) a = load_a_f16((const _Float16*)xin_, rowA, k0, lane);
        else      a = load_a_f32((const float*)xin_,    rowA, k0, lane);
#pragma unroll
        for (int q = 0; q < 4; ++q) {
            const _Float16* bp = Wih + (size_t)(q * HID + col) * 512 + k0 + bOff;
            const v8h blo = *(const v8h*)bp; const v8h bhi = *(const v8h*)(bp + 16);
            const v16h bf = __builtin_shufflevector(blo, bhi, 0,1,2,3,4,5,6,7,8,9,10,11,12,13,14,15);
            acc[q] = __builtin_amdgcn_wmma_f32_16x16x32_f16(
                false, a, false, bf, (short)0, acc[q], false, false);
        }
    }

    // ---- recurrent contribution: (m*hprev) @ Whh^T ----
#pragma unroll 4
    for (int kk = 0; kk < 16; ++kk) {
        const int k0 = kk * 32;
        const v16h a = load_a_f16_scaled(hprev, rowA, k0, lane, mh);
#pragma unroll
        for (int q = 0; q < 4; ++q) {
            const _Float16* bp = Whh + (size_t)(q * HID + col) * 512 + k0 + bOff;
            const v8h blo = *(const v8h*)bp; const v8h bhi = *(const v8h*)(bp + 16);
            const v16h bf = __builtin_shufflevector(blo, bhi, 0,1,2,3,4,5,6,7,8,9,10,11,12,13,14,15);
            acc[q] = __builtin_amdgcn_wmma_f32_16x16x32_f16(
                false, a, false, bf, (short)0, acc[q], false, false);
        }
    }

    // ---- gate nonlinearity + state update (exclusive ownership per element) ----
#pragma unroll
    for (int v = 0; v < 8; ++v) {
        const int row = bt * 16 + ((lane >> 4) << 3) + v;  // C/D row (batch)
        const float iv = sigmoidf_(acc[0][v]);
        const float fv = sigmoidf_(acc[1][v]);
        const float gv = tanhf(acc[2][v]);
        const float ov = sigmoidf_(acc[3][v]);
        const float m  = mask_t[row];
        const float cold = cstate[row * 1024 + col] * m;
        const float cnew = fv * cold + iv * gv;
        const float hnew = ov * tanhf(cnew);
        cstate[row * 1024 + col] = cnew;
        hnext[row * 512 + col]   = (_Float16)hnew;
        if (out_t)  out_t[row * 512 + col]  = hnew;
        if (hdump)  hdump[row * 1024 + col] = hnew;
    }
}

// Persistent scan: the entire T-step, 2-layer recurrence in one launch.
// 32 blocks x 256 threads = 256 waves; wave w owns batch tile w>>5 and
// hidden tile w&31. One device barrier per step (between layer0 and layer1);
// ping-pong h buffers make all other cross-step region pairs disjoint:
//   t+1 layer-0 writes hr[0,NBH) / reads hw[0,NBH);
//   t   layer-1 reads  hw[0,NBH) (read/read) and hr[NBH,2NBH) (disjoint).
template <bool XF16>
__global__ __launch_bounds__(256) void lstm_scan_kernel(
    const void*     __restrict__ x_,      // (T,128,512) f16 or fp32
    _Float16*       __restrict__ hbuf,    // 2 * (L,128,512) f16 ping-pong
    float*          __restrict__ hxs_out, // (L,128,1024) fp32 in d_out
    const _Float16* __restrict__ Wih16,   // (L,2048,512)
    const _Float16* __restrict__ Whh16,   // (L,2048,512)
    const float*    __restrict__ bias,    // (L,2048)
    const float*    __restrict__ masks,   // (T,128)
    float*          __restrict__ out,     // (T,128,512) fp32
    unsigned*       __restrict__ barrier_state) // [cnt, gen]
{
    const int lane = threadIdx.x & 31;
    const int w    = blockIdx.x * 8 + (threadIdx.x >> 5);
    const int bt   = w >> 5;   // 0..7
    const int ht   = w & 31;   // 0..31

    unsigned* cnt = barrier_state;
    unsigned* gen = barrier_state + 1;

    const size_t NW2 = (size_t)G4 * 512;  // per-layer weight elements
    float* c0 = hxs_out + HID;                           // c of layer 0
    float* c1 = hxs_out + (size_t)NB * 2 * HID + HID;    // c of layer 1

    for (int t = 0; t < T_STEPS; ++t) {
        const int p = t & 1;
        _Float16* hr = hbuf + (size_t)p * (NL * NBH);
        _Float16* hw = hbuf + (size_t)(p ^ 1) * (NL * NBH);
        const float* mt = masks + (size_t)t * NB;
        const bool last = (t == T_STEPS - 1);
        const void* xt = XF16
            ? (const void*)((const _Float16*)x_ + (size_t)t * NB * DIN)
            : (const void*)((const float*)x_    + (size_t)t * NB * DIN);

        // layer 0: input = x_t
        lstm_layer_tile<XF16>(xt, hr, hw, c0,
                              Wih16, Whh16, bias, mt,
                              nullptr, last ? hxs_out : nullptr,
                              lane, bt, ht);

        // all of fresh h0 must be visible before any wave's layer-1 GEMM
        grid_barrier(cnt, gen);

        // layer 1: input = fresh h0 (unmasked, f16)
        lstm_layer_tile<true>((const void*)hw, hr + NBH, hw + NBH, c1,
                              Wih16 + NW2, Whh16 + NW2, bias + G4, mt,
                              out + (size_t)t * NB * HID,
                              last ? hxs_out + (size_t)NB * 2 * HID : nullptr,
                              lane, bt, ht);
    }
}

// Prep: fp32->f16 weights, bias sum, copy hxs into d_out, init f16 h ping
// buffer, reset barrier state, optionally convert the x sequence to f16.
__global__ __launch_bounds__(256) void prep_kernel(
    const float* __restrict__ Wih, const float* __restrict__ Whh,
    const float* __restrict__ bih, const float* __restrict__ bhh,
    const float* __restrict__ hxs, const float* __restrict__ x,
    _Float16* __restrict__ Wih16, _Float16* __restrict__ Whh16,
    float* __restrict__ bias, float* __restrict__ hxs_out,
    _Float16* __restrict__ hbuf0, unsigned* __restrict__ barrier_state,
    _Float16* __restrict__ x16, long ntot)
{
    const long NW = (long)NL * G4 * 512;  // 2,097,152
    for (long i = (long)blockIdx.x * blockDim.x + threadIdx.x; i < ntot;
         i += (long)gridDim.x * blockDim.x) {
        if (i < 2) barrier_state[i] = 0u;
        if (i < NW) {
            Wih16[i] = (_Float16)Wih[i];
            Whh16[i] = (_Float16)Whh[i];
        }
        if (i < NL * G4) bias[i] = bih[i] + bhh[i];
        if (i < NL * NB * 2 * HID) hxs_out[i] = hxs[i];  // 262,144
        if (i < NL * NBH) {                               // 131,072
            const int l = (int)(i >> 16);
            const int r = (int)(i & 65535);
            const int n = r >> 9;
            const int k = r & 511;
            hbuf0[i] = (_Float16)hxs[l * (NB * 2 * HID) + n * (2 * HID) + k];
        }
        if (x16) x16[i] = (_Float16)x[i];
    }
}

// In-place LayerNorm over last dim (H=512). One block of 128 threads per row.
__global__ __launch_bounds__(128) void layernorm_kernel(
    float* __restrict__ out, const float* __restrict__ gamma,
    const float* __restrict__ beta)
{
    const size_t rowbase = (size_t)blockIdx.x * HID;
    float vals[4];
    float s = 0.f, s2 = 0.f;
#pragma unroll
    for (int j = 0; j < 4; ++j) {
        const float xv = out[rowbase + j * 128 + threadIdx.x];
        vals[j] = xv; s += xv; s2 += xv * xv;
    }
#pragma unroll
    for (int off = 16; off > 0; off >>= 1) {
        s  += __shfl_down(s,  off);
        s2 += __shfl_down(s2, off);
    }
    __shared__ float red[8];
    const int wv = threadIdx.x >> 5;
    if ((threadIdx.x & 31) == 0) { red[wv] = s; red[4 + wv] = s2; }
    __syncthreads();
    __shared__ float stats[2];
    if (threadIdx.x == 0) {
        const float ts = red[0] + red[1] + red[2] + red[3];
        const float tq = red[4] + red[5] + red[6] + red[7];
        const float mu  = ts * (1.0f / HID);
        const float var = tq * (1.0f / HID) - mu * mu;
        stats[0] = mu;
        stats[1] = rsqrtf(var + EPS);
    }
    __syncthreads();
    const float mu = stats[0], rstd = stats[1];
#pragma unroll
    for (int j = 0; j < 4; ++j) {
        const int c = j * 128 + threadIdx.x;
        out[rowbase + c] = (vals[j] - mu) * rstd * gamma[c] + beta[c];
    }
}

extern "C" void kernel_launch(void* const* d_in, const int* in_sizes, int n_in,
                              void* d_out, int out_size, void* d_ws, size_t ws_size,
                              hipStream_t stream) {
    (void)in_sizes; (void)n_in; (void)out_size;
    const float* x     = (const float*)d_in[0];  // (T,N,D)
    const float* hxs   = (const float*)d_in[1];  // (L,N,2H)
    const float* masks = (const float*)d_in[2];  // (T,N)
    const float* Wih   = (const float*)d_in[3];  // (L,4H,D)
    const float* Whh   = (const float*)d_in[4];  // (L,4H,H)
    const float* bih   = (const float*)d_in[5];
    const float* bhh   = (const float*)d_in[6];
    const float* gamma = (const float*)d_in[7];
    const float* beta  = (const float*)d_in[8];

    float* out     = (float*)d_out;                       // (T,N,H)
    float* hxs_out = out + (size_t)T_STEPS * NB * HID;    // (L,N,2H)

    // Workspace layout
    const size_t NW  = (size_t)NL * G4 * 512;      // weight elements (each tensor)
    const size_t HB  = (size_t)NL * NBH;           // h elements per ping buffer
    const size_t TND = (size_t)T_STEPS * NB * DIN; // x elements

    _Float16* Wih16 = (_Float16*)d_ws;                    // 4 MB
    _Float16* Whh16 = Wih16 + NW;                         // 4 MB
    float*    bias  = (float*)(Whh16 + NW);               // 16 KB
    _Float16* hbuf  = (_Float16*)(bias + NL * G4);        // 2 * 256 KB
    unsigned* bar   = (unsigned*)(hbuf + 2 * HB);         // 64 B (padded)
    _Float16* x16   = (_Float16*)((char*)bar + 64);       // 64 MB (optional)

    const size_t wsNeedBase = (size_t)((char*)x16 - (char*)d_ws);
    const bool useXF16 = (ws_size >= wsNeedBase + TND * sizeof(_Float16));
    if (!useXF16) x16 = nullptr;

    const long prepTot = useXF16 ? (long)TND : (long)NW;
    prep_kernel<<<8192, 256, 0, stream>>>(
        Wih, Whh, bih, bhh, hxs, x, Wih16, Whh16, bias, hxs_out, hbuf, bar,
        x16, prepTot);

    if (useXF16) {
        lstm_scan_kernel<true><<<NBLOCKS, 256, 0, stream>>>(
            (const void*)x16, hbuf, hxs_out, Wih16, Whh16, bias, masks, out, bar);
    } else {
        lstm_scan_kernel<false><<<NBLOCKS, 256, 0, stream>>>(
            (const void*)x, hbuf, hxs_out, Wih16, Whh16, bias, masks, out, bar);
    }

    layernorm_kernel<<<T_STEPS * NB, 128, 0, stream>>>(out, gamma, beta);
}
